// PointNetSetAbstraction_14912126451976
// MI455X (gfx1250) — matrix operand order repaired and checked by reading, not compiled, not run on hardware
//
#include <hip/hip_runtime.h>
#include <hip/hip_bf16.h>

typedef __attribute__((ext_vector_type(16))) _Float16 v16h;
typedef __attribute__((ext_vector_type(8)))  float    v8f;
typedef __attribute__((ext_vector_type(4)))  unsigned int u32x4;

#define Bx 16
#define Nx 8192
#define Sx 512
#define Kx 32
#define Mx (Bx * Sx * Kx)   // 262144 rows for the MLP GEMMs
#define R2 0.16f

union ABu { u32x4 u[2]; v16h h; };

// ---------------------------------------------------------------------------
// 1) Furthest point sampling: one workgroup per batch, dist[] lives in LDS,
//    512 dependent argmax steps. Writes centroids straight into d_out.
// ---------------------------------------------------------------------------
__global__ __launch_bounds__(256) void fps_kernel(const float* __restrict__ data,
                                                  float* __restrict__ cent_out) {
    const int b = blockIdx.x;
    const int tid = threadIdx.x;
    __shared__ float sdist[Nx];
    __shared__ float rv[256];
    __shared__ int   ri[256];
    __shared__ int   sfar;
    const float* pd = data + (size_t)b * Nx * 3;

    for (int i = tid; i < Nx; i += 256) sdist[i] = 1e10f;
    if (tid == 0) sfar = 0;
    __syncthreads();

    for (int it = 0; it < Sx; ++it) {
        const int far = sfar;
        const float cx = pd[far * 3 + 0];
        const float cy = pd[far * 3 + 1];
        const float cz = pd[far * 3 + 2];
        if (tid == 0) {
            float* c = cent_out + ((size_t)b * Sx + it) * 3;
            c[0] = cx; c[1] = cy; c[2] = cz;
        }
        float best = -1.0f; int bidx = 0;
        for (int i = tid; i < Nx; i += 256) {
            const float dx = pd[i * 3 + 0] - cx;
            const float dy = pd[i * 3 + 1] - cy;
            const float dz = pd[i * 3 + 2] - cz;
            const float d  = dx * dx + dy * dy + dz * dz;
            const float nd = fminf(sdist[i], d);
            sdist[i] = nd;
            if (nd > best) { best = nd; bidx = i; }
        }
        rv[tid] = best; ri[tid] = bidx;
        __syncthreads();
        for (int off = 128; off > 0; off >>= 1) {
            if (tid < off) {
                const float ov = rv[tid + off]; const int oi = ri[tid + off];
                if (ov > rv[tid] || (ov == rv[tid] && oi < ri[tid])) { rv[tid] = ov; ri[tid] = oi; }
            }
            __syncthreads();
        }
        if (tid == 0) sfar = ri[0];
        __syncthreads();
    }
}

// ---------------------------------------------------------------------------
// 2) Ball query: one wave32 per query; append first K in-radius indices in
//    index order via ballot + prefix popcount; pad with first hit.
// ---------------------------------------------------------------------------
__global__ __launch_bounds__(256) void ballq_kernel(const float* __restrict__ data,
                                                    const float* __restrict__ cent,
                                                    int* __restrict__ nbr) {
    const int wave = (int)((blockIdx.x * blockDim.x + threadIdx.x) >> 5);  // 0..8191
    const int lane = threadIdx.x & 31;
    const int wib  = threadIdx.x >> 5;
    __shared__ int snbr[8][Kx];
    int* my = snbr[wib];

    const int b = wave >> 9;
    const float qx = cent[wave * 3 + 0];
    const float qy = cent[wave * 3 + 1];
    const float qz = cent[wave * 3 + 2];
    const float* pd = data + (size_t)b * Nx * 3;

    int cnt = 0;
    for (int base = 0; base < Nx && cnt < Kx; base += 32) {
        const int i = base + lane;
        const float dx = pd[i * 3 + 0] - qx;
        const float dy = pd[i * 3 + 1] - qy;
        const float dz = pd[i * 3 + 2] - qz;
        const bool ok = (dx * dx + dy * dy + dz * dz) <= R2;
        const unsigned mask = (unsigned)__ballot(ok);
        if (ok) {
            const int pos = cnt + __popc(mask & ((1u << lane) - 1u));
            if (pos < Kx) my[pos] = i;
        }
        cnt += __popc(mask);
    }
    __syncthreads();
    int v;
    if (cnt == 0) v = 0;
    else          v = my[lane < cnt ? lane : 0];
    nbr[(size_t)wave * Kx + lane] = v;
}

// ---------------------------------------------------------------------------
// 3) Build layer-0 A matrix (f16, K padded 6->32): gather + center + concat.
// ---------------------------------------------------------------------------
__global__ __launch_bounds__(256) void prep0_kernel(const float* __restrict__ data,
                                                    const float* __restrict__ feat,
                                                    const float* __restrict__ cent,
                                                    const int*   __restrict__ nbr,
                                                    _Float16*    __restrict__ A0) {
    const int m  = blockIdx.x * 256 + threadIdx.x;    // 0..Mx
    const int bs = m >> 5;                            // (b,s) flat
    const int b  = m >> 14;                           // m / (K*S)
    const int i  = nbr[m];
    const float* pp = data + ((size_t)b * Nx + i) * 3;
    const float* pf = feat + ((size_t)b * Nx + i) * 3;
    const float* pc = cent + (size_t)bs * 3;
    _Float16* row = A0 + (size_t)m * 32;
    row[0] = (_Float16)(pp[0] - pc[0]);
    row[1] = (_Float16)(pp[1] - pc[1]);
    row[2] = (_Float16)(pp[2] - pc[2]);
    row[3] = (_Float16)pf[0];
    row[4] = (_Float16)pf[1];
    row[5] = (_Float16)pf[2];
    #pragma unroll
    for (int c = 6; c < 32; ++c) row[c] = (_Float16)0.0f;
}

// ---------------------------------------------------------------------------
// 4) Convert weights to f16 (layer0 K-padded to 32).
// ---------------------------------------------------------------------------
__global__ __launch_bounds__(256) void wconv_kernel(const float* __restrict__ w0,
                                                    const float* __restrict__ w1,
                                                    const float* __restrict__ w2,
                                                    _Float16* __restrict__ W0h,
                                                    _Float16* __restrict__ W1h,
                                                    _Float16* __restrict__ W2h) {
    const int t = threadIdx.x;
    for (int idx = t; idx < 64 * 32; idx += 256) {
        const int o = idx >> 5, c = idx & 31;
        W0h[idx] = (_Float16)(c < 6 ? w0[o * 6 + c] : 0.0f);
    }
    for (int idx = t; idx < 64 * 64; idx += 256)  W1h[idx] = (_Float16)w1[idx];
    for (int idx = t; idx < 128 * 64; idx += 256) W2h[idx] = (_Float16)w2[idx];
}

// ---------------------------------------------------------------------------
// 5) WMMA GEMM: Y[M][NOUT] = A[M][KD] * W[NOUT][KD]^T + bias.
//    Block = 8 waves = 128 rows; each wave owns a 16-row strip, loops N tiles.
//    v_wmma_f32_16x16x32_f16, f32 accumulate.
// ---------------------------------------------------------------------------
template <int KD, int NOUT>
__global__ __launch_bounds__(256) void gemm_kernel(const _Float16* __restrict__ A,
                                                   const _Float16* __restrict__ W,
                                                   const float*    __restrict__ bias,
                                                   float*          __restrict__ Y) {
    __shared__ __align__(16) _Float16 As[128 * KD];
    __shared__ __align__(16) _Float16 Ws[NOUT * KD];
    const int tid = threadIdx.x;
    const size_t blockRow = (size_t)blockIdx.x * 128;

    // Stage A tile and full W into LDS with 16B vector copies.
    {
        const u32x4* gA = (const u32x4*)(A + blockRow * KD);
        u32x4* sA = (u32x4*)As;
        constexpr int AV = 128 * KD / 8;
        #pragma unroll
        for (int i = 0; i < AV / 256; ++i) sA[tid + i * 256] = gA[tid + i * 256];

        const u32x4* gW = (const u32x4*)W;
        u32x4* sW = (u32x4*)Ws;
        constexpr int WV = NOUT * KD / 8;
        for (int i = tid; i < WV; i += 256) sW[i] = gW[i];
    }
    __syncthreads();

    const int wid  = tid >> 5;
    const int lane = tid & 31;
    const int sel  = lane >> 4;       // lane half selects K sub-block
    const int l16  = lane & 15;
    const int rowBase = wid * 16;
    const int arow = rowBase + l16;   // A-matrix row for this lane

    for (int nt = 0; nt < NOUT / 16; ++nt) {
        v8f acc = {};
        #pragma unroll
        for (int kt = 0; kt < KD / 32; ++kt) {
            ABu a, bm;
            // A frag: lanes 0-15 hold K {0..7,16..23}, lanes 16-31 hold {8..15,24..31}
            a.u[0] = *(const u32x4*)&As[arow * KD + kt * 32 + sel * 8];
            a.u[1] = *(const u32x4*)&As[arow * KD + kt * 32 + 16 + sel * 8];
            // B frag: B[k][n] = W[n][k]; same K order as A: {0..7,16..23} / {8..15,24..31}
            const u32x4* bp = (const u32x4*)&Ws[(nt * 16 + l16) * KD + kt * 32 + sel * 8];
            bm.u[0] = bp[0]; bm.u[1] = bp[2];
            acc = __builtin_amdgcn_wmma_f32_16x16x32_f16(
                false, a.h, false, bm.h, (short)0, acc, false, false);
        }
        // D layout: VGPR j -> row j (lanes 0-15) / row 8+j (lanes 16-31), col = lane%16
        const size_t gr  = blockRow + rowBase + sel * 8;
        const int    col = nt * 16 + l16;
        const float  bv  = bias[col];
        #pragma unroll
        for (int j = 0; j < 8; ++j)
            Y[(gr + j) * NOUT + col] = acc[j] + bv;
    }
}

// ---------------------------------------------------------------------------
// 6) BatchNorm batch statistics: deterministic two-stage reduction.
// ---------------------------------------------------------------------------
template <int NOUT>
__global__ void bn_partial_kernel(const float* __restrict__ Y,
                                  float* __restrict__ psum, float* __restrict__ psq) {
    const int c = threadIdx.x;                 // NOUT threads
    const size_t base = (size_t)blockIdx.x * (Mx / 256);
    float s = 0.0f, q = 0.0f;
    for (int r = 0; r < Mx / 256; ++r) {
        const float y = Y[(base + r) * NOUT + c];
        s += y; q += y * y;
    }
    psum[blockIdx.x * NOUT + c] = s;
    psq [blockIdx.x * NOUT + c] = q;
}

template <int NOUT>
__global__ void bn_final_kernel(const float* __restrict__ psum, const float* __restrict__ psq,
                                const float* __restrict__ g, const float* __restrict__ be,
                                float* __restrict__ scale, float* __restrict__ shift) {
    const int c = threadIdx.x;
    float s = 0.0f, q = 0.0f;
    for (int i = 0; i < 256; ++i) { s += psum[i * NOUT + c]; q += psq[i * NOUT + c]; }
    const float m   = s * (1.0f / (float)Mx);
    const float v   = q * (1.0f / (float)Mx) - m * m;
    const float inv = rsqrtf(v + 1e-5f);
    const float a   = g[c] * inv;
    scale[c] = a;
    shift[c] = be[c] - m * a;
}

// 7) Apply BN + ReLU, emit f16 A matrix for the next layer.
template <int NOUT>
__global__ __launch_bounds__(256) void act_kernel(const float* __restrict__ Y,
                                                  const float* __restrict__ scale,
                                                  const float* __restrict__ shift,
                                                  _Float16* __restrict__ Aout) {
    const size_t idx = (size_t)blockIdx.x * 256 + threadIdx.x;
    const int c = (int)(idx & (NOUT - 1));
    const float y = fmaxf(Y[idx] * scale[c] + shift[c], 0.0f);
    Aout[idx] = (_Float16)y;
}

// 8) Final layer BN + ReLU + max over K (relu commutes with max).
__global__ __launch_bounds__(128) void final_kernel(const float* __restrict__ Y2,
                                                    const float* __restrict__ scale,
                                                    const float* __restrict__ shift,
                                                    float* __restrict__ out) {
    const int bs = blockIdx.x;     // 8192
    const int c  = threadIdx.x;    // 128
    const float a = scale[c], sh = shift[c];
    float mp = -1e30f;
    const float* base = Y2 + ((size_t)bs * Kx) * 128 + c;
    #pragma unroll 4
    for (int k = 0; k < Kx; ++k)
        mp = fmaxf(mp, base[(size_t)k * 128] * a + sh);
    out[(size_t)bs * 128 + c] = fmaxf(mp, 0.0f);
}

// ---------------------------------------------------------------------------
extern "C" void kernel_launch(void* const* d_in, const int* in_sizes, int n_in,
                              void* d_out, int out_size, void* d_ws, size_t ws_size,
                              hipStream_t stream) {
    const float* data = (const float*)d_in[0];
    const float* feat = (const float*)d_in[1];
    const float* w0 = (const float*)d_in[2];
    const float* b0 = (const float*)d_in[3];
    const float* g0 = (const float*)d_in[4];
    const float* be0 = (const float*)d_in[5];
    const float* w1 = (const float*)d_in[6];
    const float* b1 = (const float*)d_in[7];
    const float* g1 = (const float*)d_in[8];
    const float* be1 = (const float*)d_in[9];
    const float* w2 = (const float*)d_in[10];
    const float* b2 = (const float*)d_in[11];
    const float* g2 = (const float*)d_in[12];
    const float* be2 = (const float*)d_in[13];

    float* cent = (float*)d_out;                       // [B,S,3]
    float* outp = (float*)d_out + (size_t)Bx * Sx * 3; // [B,S,128]

    char* ws = (char*)d_ws;
    size_t off = 0;
    auto take = [&](size_t bytes) -> void* {
        void* p = ws + off;
        off += (bytes + 255) & ~(size_t)255;
        return p;
    };
    int*      nbr  = (int*)     take((size_t)Mx * sizeof(int));
    _Float16* Abuf = (_Float16*)take((size_t)Mx * 64 * sizeof(_Float16));  // also holds [Mx][32]
    float*    Ybuf = (float*)   take((size_t)Mx * 128 * sizeof(float));
    _Float16* W0h  = (_Float16*)take(64 * 32 * sizeof(_Float16));
    _Float16* W1h  = (_Float16*)take(64 * 64 * sizeof(_Float16));
    _Float16* W2h  = (_Float16*)take(128 * 64 * sizeof(_Float16));
    float*    psum = (float*)   take(256 * 128 * sizeof(float));
    float*    psq  = (float*)   take(256 * 128 * sizeof(float));
    float*    scl  = (float*)   take(128 * sizeof(float));
    float*    shf  = (float*)   take(128 * sizeof(float));

    // Stage 1: sampling + grouping
    fps_kernel<<<Bx, 256, 0, stream>>>(data, cent);
    ballq_kernel<<<(Bx * Sx) / 8, 256, 0, stream>>>(data, cent, nbr);
    wconv_kernel<<<1, 256, 0, stream>>>(w0, w1, w2, W0h, W1h, W2h);
    prep0_kernel<<<Mx / 256, 256, 0, stream>>>(data, feat, cent, nbr, Abuf);

    // Layer 0: 6(->32) -> 64
    gemm_kernel<32, 64><<<Mx / 128, 256, 0, stream>>>(Abuf, W0h, b0, Ybuf);
    bn_partial_kernel<64><<<256, 64, 0, stream>>>(Ybuf, psum, psq);
    bn_final_kernel<64><<<1, 64, 0, stream>>>(psum, psq, g0, be0, scl, shf);
    act_kernel<64><<<(size_t)Mx * 64 / 256, 256, 0, stream>>>(Ybuf, scl, shf, Abuf);

    // Layer 1: 64 -> 64
    gemm_kernel<64, 64><<<Mx / 128, 256, 0, stream>>>(Abuf, W1h, b1, Ybuf);
    bn_partial_kernel<64><<<256, 64, 0, stream>>>(Ybuf, psum, psq);
    bn_final_kernel<64><<<1, 64, 0, stream>>>(psum, psq, g1, be1, scl, shf);
    act_kernel<64><<<(size_t)Mx * 64 / 256, 256, 0, stream>>>(Ybuf, scl, shf, Abuf);

    // Layer 2: 64 -> 128
    gemm_kernel<64, 128><<<Mx / 128, 256, 0, stream>>>(Abuf, W2h, b2, Ybuf);
    bn_partial_kernel<128><<<256, 128, 0, stream>>>(Ybuf, psum, psq);
    bn_final_kernel<128><<<1, 128, 0, stream>>>(psum, psq, g2, be2, scl, shf);

    // BN + ReLU + max over K
    final_kernel<<<Bx * Sx, 128, 0, stream>>>(Ybuf, scl, shf, outp);
}
